// EquivariantUpdate_45973329936455
// MI455X (gfx1250) — matrix operand order repaired and checked by reading, not compiled, not run on hardware
//
#include <hip/hip_runtime.h>
#include <math.h>

// ---------------- problem constants ----------------
#define N_ATOMS 50000
#define N_EDGES 500000
#define N_FRAG  2000
#define KPAD    2016      // F padded up to a multiple of 32 for the K loop
#define BCOLS   144       // 128 MLP cols + 3 coord cols + pad to 9 wmma tiles
#define CRANGE  10.0f     // COORDS_RANGE

typedef __attribute__((ext_vector_type(16))) __bf16 v16bf;
typedef __attribute__((ext_vector_type(8)))  float  v8f;

union Frag { unsigned int u[8]; v16bf v; };

__device__ __forceinline__ unsigned int f2bf(float x) {
    union { float f; unsigned int u; } v; v.f = x;
    unsigned int r = v.u + 0x7fffu + ((v.u >> 16) & 1u);   // round-to-nearest-even
    return r >> 16;
}
__device__ __forceinline__ float bf2f(unsigned short h) {
    union { unsigned int u; float f; } v; v.u = ((unsigned int)h) << 16; return v.f;
}
__device__ __forceinline__ float silu_f(float x) { return x / (1.0f + __expf(-x)); }

// pack 8 consecutive f32 -> 8 bf16 in a uint4 (for ds_store_b128 staging)
__device__ __forceinline__ uint4 pack8(const float* __restrict__ p) {
    float4 a = *(const float4*)p;
    float4 b = *(const float4*)(p + 4);
    uint4 r;
    r.x = f2bf(a.x) | (f2bf(a.y) << 16);
    r.y = f2bf(a.z) | (f2bf(a.w) << 16);
    r.z = f2bf(b.x) | (f2bf(b.y) << 16);
    r.w = f2bf(b.z) | (f2bf(b.w) << 16);
    return r;
}

// A fragment: 16x32 bf16 tile, rows in LDS with row stride rs (elements).
// ISA 7.12.2: lane&15 = M; lanes 0-15 hold K 0-7,16-23; lanes 16-31 hold K 8-15,24-31.
__device__ __forceinline__ void load_a(Frag& a, const unsigned short* S, int m,
                                       int kbase, int rs, int hi) {
    const unsigned short* p = S + m * rs + kbase + hi * 8;
#pragma unroll
    for (int j = 0; j < 4; ++j) a.u[j]     = *(const unsigned int*)(p + 2 * j);
#pragma unroll
    for (int j = 0; j < 4; ++j) a.u[4 + j] = *(const unsigned int*)(p + 16 + 2 * j);
}
// B fragment: 32x16 bf16 (KxN) from K-major (pre-transposed) weights, row stride ks.
// lane&15 = N; lanes 0-15 hold K 0-7,16-23; lanes 16-31 hold K 8-15,24-31.
__device__ __forceinline__ void load_b(Frag& b, const unsigned short* WT, int n,
                                       int kbase, int ks, int hi) {
    const unsigned short* p = WT + n * ks + kbase + hi * 8;
#pragma unroll
    for (int j = 0; j < 4; ++j) b.u[j]     = *(const unsigned int*)(p + 2 * j);
#pragma unroll
    for (int j = 0; j < 4; ++j) b.u[4 + j] = *(const unsigned int*)(p + 16 + 2 * j);
}

#define WMMA_BF16(ACC, A, B) \
    (ACC) = __builtin_amdgcn_wmma_f32_16x16x32_bf16(false, (A).v, false, (B).v, (short)0, (ACC), false, false)

// uniform vectorized stage of one 16x32 f32 bm tile -> bf16 LDS tile (256 threads)
__device__ __forceinline__ void stage_bm(const float* __restrict__ bm,
                                         unsigned short* __restrict__ dst,
                                         int m0, int kb, int t) {
    if (t < 256) {
        int r = t >> 4, c = (t & 15) * 2;
        unsigned int packed = 0;
        if (kb + c < N_FRAG) {                       // false only in the K tail
            const float2 v = *(const float2*)(bm + (size_t)(m0 + r) * N_FRAG + kb + c);
            packed = f2bf(v.x) | (f2bf(v.y) << 16);
        }
        *(unsigned int*)(dst + r * 32 + c) = packed;
    }
}

// ---------------- prep kernels (tiny) ----------------
__global__ void k_convert(const float* __restrict__ aW1, const float* __restrict__ aW2,
                          const float* __restrict__ fW1, const float* __restrict__ fW2,
                          unsigned short* __restrict__ w1aT, unsigned short* __restrict__ w1bT,
                          unsigned short* __restrict__ aw2T, unsigned short* __restrict__ fw1aT,
                          unsigned short* __restrict__ fw2T,
                          float* __restrict__ awAttr, float* __restrict__ fwRad) {
    int idx = blockIdx.x * blockDim.x + threadIdx.x;
    if (idx < 5 * 16384) {
        int mat = idx >> 14, r = idx & 16383;
        int nn = r >> 7, kk = r & 127;
        float v; unsigned short* dst;
        switch (mat) {
            case 0:  v = aW1[kk * 128 + nn];         dst = w1aT;  break;
            case 1:  v = aW1[(128 + kk) * 128 + nn]; dst = w1bT;  break;
            case 2:  v = aW2[kk * 128 + nn];         dst = aw2T;  break;
            case 3:  v = fW1[kk * 128 + nn];         dst = fw1aT; break;
            default: v = fW2[kk * 128 + nn];         dst = fw2T;  break;
        }
        dst[nn * 128 + kk] = (unsigned short)f2bf(v);
    } else if (idx < 5 * 16384 + 256) {
        int r = idx - 5 * 16384;           // r = j*128 + n, j in {0,1}
        int j = r >> 7, nn = r & 127;
        awAttr[r] = aW1[(256 + j) * 128 + nn];
    } else if (idx < 5 * 16384 + 256 + 128) {
        int nn = idx - (5 * 16384 + 256);
        fwRad[nn] = fW1[256 * 128 + nn] + fW1[257 * 128 + nn];
    }
}

// bond_proj[type][n] = bond_emb[type]@aW1[258:386] + ab1  (100x128, f32)
__global__ void k_bondproj(const float* __restrict__ bemb, const float* __restrict__ aW1,
                           const float* __restrict__ ab1, float* __restrict__ bondp) {
    int idx = blockIdx.x * blockDim.x + threadIdx.x;
    if (idx >= 100 * 128) return;
    int ty = idx >> 7, nn = idx & 127;
    float s = ab1[nn];
    for (int k = 0; k < 128; ++k) s += bemb[ty * 128 + k] * aW1[(258 + k) * 128 + nn];
    bondp[idx] = s;
}

// hfBT[n][f] (K-major, bf16, 144 x 2016): rows 0-127 = (h_f @ fW1[128:256])^T,
// rows 128-130 = x_f^T, rest 0.  Folds (bm@h_f)@fW1b = bm@(h_f@fW1b).
__global__ void k_hfbt(const float* __restrict__ h_f, const float* __restrict__ fW1,
                       const float* __restrict__ x_f, unsigned short* __restrict__ hfBT) {
    int idx = blockIdx.x * blockDim.x + threadIdx.x;
    if (idx >= BCOLS * KPAD) return;
    int nn = idx / KPAD, f = idx % KPAD;
    float v = 0.0f;
    if (f < N_FRAG) {
        if (nn < 128) {
            for (int k = 0; k < 128; ++k) v += h_f[f * 128 + k] * fW1[(128 + k) * 128 + nn];
        } else if (nn < 131) {
            v = x_f[f * 3 + (nn - 128)];
        }
    }
    hfBT[idx] = (unsigned short)f2bf(v);
}

// ---------------- fragment path: big GEMM fused with the fragment MLP ----------------
// 9 waves / block; waves 0-7 own the 128 MLP columns, wave 8 the coord columns.
__global__ __launch_bounds__(288) void k_frag(
    const float* __restrict__ bm, const float* __restrict__ h_a,
    const float* __restrict__ x_a, const float* __restrict__ fb1,
    const float* __restrict__ fb2, const float* __restrict__ fW3,
    const unsigned short* __restrict__ hfBT, const unsigned short* __restrict__ fw1aT,
    const unsigned short* __restrict__ fw2T, const float* __restrict__ fwRad,
    float* __restrict__ out) {
    __shared__ __attribute__((aligned(16))) unsigned short sA0[16 * 32];
    __shared__ __attribute__((aligned(16))) unsigned short sA1[16 * 32];
    __shared__ __attribute__((aligned(16))) unsigned short sH[16 * 128];
    __shared__ __attribute__((aligned(16))) unsigned short sU[16 * 128];
    __shared__ float sBX[16 * 3];
    __shared__ float sRad[16];
    __shared__ float sCdn[16 * 3];

    const int t = threadIdx.x;
    const int w = t >> 5;            // 0..8
    const int lane = t & 31;
    const int hi = lane >> 4;
    const int nl = lane & 15;
    const int m0 = blockIdx.x * 16;  // 3125 * 16 = 50000

    // stage h_a tile (bf16, vectorized: 8 elements/thread, ds_store_b128)
    if (t < 256) {
        int r = t >> 4, k = (t & 15) * 8;
        *(uint4*)(sH + r * 128 + k) = pack8(h_a + (size_t)(m0 + r) * 128 + k);
    }

    // big GEMM: acc(ntile w) = bm[m0:m0+16, 0:2000] @ [hf_proj | x_f]
    // double-buffered LDS staging: one barrier per K step, next tile's global
    // load issued before the current WMMA.
    v8f acc = {};
    stage_bm(bm, sA0, m0, 0, t);
    for (int s = 0; s < 63; ++s) {
        __syncthreads();
        const unsigned short* cur = (s & 1) ? sA1 : sA0;
        if (s + 1 < 63) stage_bm(bm, (s & 1) ? sA0 : sA1, m0, (s + 1) * 32, t);
        Frag a, b;
        load_a(a, cur, nl, 0, 32, hi);
        load_b(b, hfBT, w * 16 + nl, s * 32, KPAD, hi);
        WMMA_BF16(acc, a, b);
    }
    __syncthreads();
    if (w == 8 && nl < 3) {          // expose bm@x_f (cols 128..130)
#pragma unroll
        for (int g = 0; g < 8; ++g) sBX[(g + hi * 8) * 3 + nl] = acc[g];
    }
    __syncthreads();
    if (t < 16) {                    // radial + normalized coord_diff_f
        float r2 = 0.0f, cd[3];
#pragma unroll
        for (int c = 0; c < 3; ++c) {
            cd[c] = x_a[(m0 + t) * 3 + c] - sBX[t * 3 + c];
            r2 += cd[c] * cd[c];
        }
        sRad[t] = r2;
        float nf = sqrtf(r2 + 1e-8f) + 1.0f;
#pragma unroll
        for (int c = 0; c < 3; ++c) sCdn[t * 3 + c] = cd[c] / nf;
    }
    __syncthreads();

    // layer 1: + h_a@fW1a + radial*(w256+w257) + fb1, silu
    if (w < 8) {
        int n = w * 16 + nl;
#pragma unroll
        for (int s = 0; s < 4; ++s) {
            Frag a, b;
            load_a(a, sH, nl, s * 32, 128, hi);
            load_b(b, fw1aT, n, s * 32, 128, hi);
            WMMA_BF16(acc, a, b);
        }
        float wr = fwRad[n], b1 = fb1[n];
#pragma unroll
        for (int g = 0; g < 8; ++g) {
            int m = g + hi * 8;
            sU[m * 128 + n] = (unsigned short)f2bf(silu_f(acc[g] + sRad[m] * wr + b1));
        }
    }
    __syncthreads();
    // layer 2
    v8f acc2 = {};
    if (w < 8) {
#pragma unroll
        for (int s = 0; s < 4; ++s) {
            Frag a, b;
            load_a(a, sU, nl, s * 32, 128, hi);
            load_b(b, fw2T, w * 16 + nl, s * 32, 128, hi);
            WMMA_BF16(acc2, a, b);
        }
    }
    __syncthreads();                 // all reads of sU done before overwrite
    if (w < 8) {
        int n = w * 16 + nl;
        float b2 = fb2[n];
#pragma unroll
        for (int g = 0; g < 8; ++g) {
            int m = g + hi * 8;
            sU[m * 128 + n] = (unsigned short)f2bf(silu_f(acc2[g] + b2));
        }
    }
    __syncthreads();
    // layer 3: parallel 8-elem partials + shfl reduction within 16-lane groups
    if (t < 256) {
        int m = t >> 4, j = t & 15, kk = j * 8;
        float part = 0.0f;
#pragma unroll
        for (int q = 0; q < 8; ++q)
            part += bf2f(sU[m * 128 + kk + q]) * fW3[kk + q];
#pragma unroll
        for (int mask = 8; mask >= 1; mask >>= 1)
            part += __shfl_xor(part, mask, 32);
        if (j == 0) {
            float tf = tanhf(part) * CRANGE;
#pragma unroll
            for (int c = 0; c < 3; ++c)
                out[(m0 + m) * 3 + c] = x_a[(m0 + m) * 3 + c] + sCdn[m * 3 + c] * tf;
        }
    }
}

// ---------------- edge path: fused 3-layer MLP + tanh + scatter-add ----------------
__global__ __launch_bounds__(256) void k_edge(
    const float* __restrict__ h_a, const int* __restrict__ eidx,
    const int* __restrict__ etype, const float* __restrict__ eattr,
    const float* __restrict__ cda, const float* __restrict__ bondp,
    const float* __restrict__ awAttr, const float* __restrict__ ab2,
    const float* __restrict__ aW3,
    const unsigned short* __restrict__ w1aT, const unsigned short* __restrict__ w1bT,
    const unsigned short* __restrict__ aw2T, float* __restrict__ out) {
    __shared__ __attribute__((aligned(16))) unsigned short sR[16 * 128];
    __shared__ __attribute__((aligned(16))) unsigned short sC[16 * 128];
    __shared__ __attribute__((aligned(16))) unsigned short sU[16 * 128];
    __shared__ int   sRow[16];
    __shared__ int   sCol[16];
    __shared__ int   sTyp[16];
    __shared__ float sE0[16];
    __shared__ float sE1[16];

    const int t = threadIdx.x;
    const int w = t >> 5;            // 0..7
    const int lane = t & 31;
    const int hi = lane >> 4;
    const int nl = lane & 15;
    const int e0 = blockIdx.x * 16;  // 31250 * 16 = 500000

    if (t < 16) {
        int e = e0 + t;
        sRow[t] = eidx[e];
        sCol[t] = eidx[N_EDGES + e];
        sTyp[t] = etype[e];
        sE0[t]  = eattr[e * 2 + 0];
        sE1[t]  = eattr[e * 2 + 1];
    }
    __syncthreads();
    {   // gather h_a rows (L2-resident), vectorized: 8 elems/thread per matrix
        int r = t >> 4, k = (t & 15) * 8;
        *(uint4*)(sR + r * 128 + k) = pack8(h_a + (size_t)sRow[r] * 128 + k);
        *(uint4*)(sC + r * 128 + k) = pack8(h_a + (size_t)sCol[r] * 128 + k);
    }
    __syncthreads();

    const int n = w * 16 + nl;
    // layer 1: h_row@W1a + h_col@W1b  (K=256 total)
    v8f acc = {};
#pragma unroll
    for (int s = 0; s < 4; ++s) {
        Frag a, b;
        load_a(a, sR, nl, s * 32, 128, hi);
        load_b(b, w1aT, n, s * 32, 128, hi);
        WMMA_BF16(acc, a, b);
    }
#pragma unroll
    for (int s = 0; s < 4; ++s) {
        Frag a, b;
        load_a(a, sC, nl, s * 32, 128, hi);
        load_b(b, w1bT, n, s * 32, 128, hi);
        WMMA_BF16(acc, a, b);
    }
    float wa0 = awAttr[n], wa1 = awAttr[128 + n];
#pragma unroll
    for (int g = 0; g < 8; ++g) {
        int m = g + hi * 8;
        float v = acc[g] + sE0[m] * wa0 + sE1[m] * wa1 + bondp[sTyp[m] * 128 + n];
        sU[m * 128 + n] = (unsigned short)f2bf(silu_f(v));
    }
    __syncthreads();
    // layer 2
    v8f acc2 = {};
#pragma unroll
    for (int s = 0; s < 4; ++s) {
        Frag a, b;
        load_a(a, sU, nl, s * 32, 128, hi);
        load_b(b, aw2T, n, s * 32, 128, hi);
        WMMA_BF16(acc2, a, b);
    }
    __syncthreads();                 // all sU reads complete
    {
        float b2 = ab2[n];
#pragma unroll
        for (int g = 0; g < 8; ++g) {
            int m = g + hi * 8;
            sU[m * 128 + n] = (unsigned short)f2bf(silu_f(acc2[g] + b2));
        }
    }
    __syncthreads();
    // layer 3: parallel 8-elem partials + shfl reduction, then tanh + f32 atomics
    {
        int m = t >> 4, j = t & 15, kk = j * 8;
        float part = 0.0f;
#pragma unroll
        for (int q = 0; q < 8; ++q)
            part += bf2f(sU[m * 128 + kk + q]) * aW3[kk + q];
#pragma unroll
        for (int mask = 8; mask >= 1; mask >>= 1)
            part += __shfl_xor(part, mask, 32);
        if (j == 0) {
            int e = e0 + m;
            float sc = tanhf(part) * (CRANGE / 100.0f);
#pragma unroll
            for (int c = 0; c < 3; ++c)
                atomicAdd(&out[sRow[m] * 3 + c], cda[e * 3 + c] * sc);
        }
    }
}

// ---------------- host entry ----------------
extern "C" void kernel_launch(void* const* d_in, const int* in_sizes, int n_in,
                              void* d_out, int out_size, void* d_ws, size_t ws_size,
                              hipStream_t stream) {
    (void)in_sizes; (void)n_in; (void)out_size; (void)ws_size;
    const float* h_a   = (const float*)d_in[0];
    const float* x_a   = (const float*)d_in[1];
    const int*   eidx  = (const int*)d_in[2];
    const int*   etype = (const int*)d_in[3];
    const float* eattr = (const float*)d_in[4];
    const float* cda   = (const float*)d_in[5];
    const float* h_f   = (const float*)d_in[6];
    const float* x_f   = (const float*)d_in[7];
    const float* bm    = (const float*)d_in[8];
    const float* bemb  = (const float*)d_in[9];
    const float* aW1   = (const float*)d_in[10];
    const float* ab1   = (const float*)d_in[11];
    const float* aW2   = (const float*)d_in[12];
    const float* ab2   = (const float*)d_in[13];
    const float* aW3   = (const float*)d_in[14];
    const float* fW1   = (const float*)d_in[15];
    const float* fb1   = (const float*)d_in[16];
    const float* fW2   = (const float*)d_in[17];
    const float* fb2   = (const float*)d_in[18];
    const float* fW3   = (const float*)d_in[19];
    float* out = (float*)d_out;

    // workspace layout (~0.78 MB, fully rewritten every call)
    char* ws = (char*)d_ws;
    unsigned short* w1aT  = (unsigned short*)(ws + 0);
    unsigned short* w1bT  = (unsigned short*)(ws + 32768);
    unsigned short* aw2T  = (unsigned short*)(ws + 65536);
    unsigned short* fw1aT = (unsigned short*)(ws + 98304);
    unsigned short* fw2T  = (unsigned short*)(ws + 131072);
    float*          awAttr= (float*)(ws + 163840);
    float*          fwRad = (float*)(ws + 164864);
    float*          bondp = (float*)(ws + 165376);
    unsigned short* hfBT  = (unsigned short*)(ws + 216576);

    k_convert<<<(5 * 16384 + 256 + 128 + 255) / 256, 256, 0, stream>>>(
        aW1, aW2, fW1, fW2, w1aT, w1bT, aw2T, fw1aT, fw2T, awAttr, fwRad);
    k_bondproj<<<(100 * 128 + 255) / 256, 256, 0, stream>>>(bemb, aW1, ab1, bondp);
    k_hfbt<<<(BCOLS * KPAD + 255) / 256, 256, 0, stream>>>(h_f, fW1, x_f, hfBT);
    k_frag<<<N_ATOMS / 16, 288, 0, stream>>>(
        bm, h_a, x_a, fb1, fb2, fW3, hfBT, fw1aT, fw2T, fwRad, out);
    k_edge<<<N_EDGES / 16, 256, 0, stream>>>(
        h_a, eidx, etype, eattr, cda, bondp, awAttr, ab2, aW3, w1aT, w1bT, aw2T, out);
}
